// NFSparseVAE_34376918237651
// MI455X (gfx1250) — matrix operand order repaired and checked by reading, not compiled, not run on hardware
//
#include <hip/hip_runtime.h>
#include <hip/hip_bf16.h>

// ---------------------------------------------------------------------------
// NFSparseVAE forward for MI455X (gfx1250, wave32, WMMA).
// Convs use v_wmma_f32_16x16x32_f16 implicit GEMM (M=16 pixels x N=16 chans,
// K = 9*Cin padded to 32). A/B tiles staged in LDS in fragment-friendly order
// so per-lane fragment loads are contiguous ds_load_b128s. Everything else
// (pools, flow, invconv) is VALU — the net is HBM-bound (~400MB traffic
// @ 23.3TB/s >> ~3 GFLOP of conv math).
// ---------------------------------------------------------------------------

typedef __attribute__((ext_vector_type(16))) _Float16 v16h;
typedef __attribute__((ext_vector_type(8)))  _Float16 v8h;
typedef __attribute__((ext_vector_type(8)))  float    v8f;

#define CDIV(a, b) (((a) + (b) - 1) / (b))

// ------------------------- problem constants -------------------------------
static constexpr int BN  = 128;
static constexpr int H0 = 200, W0 = 128;   // mask m0 res
static constexpr int H1 = 100, W1 = 64;    // m1
static constexpr int H2 = 20,  W2 = 32;    // m2
static constexpr int H3 = 4,   W3 = 16;    // m3
static constexpr int LATD = 4, HIDD = 16, NPP = 11, NFLOW = 10;

__device__ __forceinline__ unsigned pack2h(float a, float b) {
  union { _Float16 h[2]; unsigned u; } t;
  t.h[0] = (_Float16)a;
  t.h[1] = (_Float16)b;
  return t.u;
}

// ---------------------------- mask from x ----------------------------------
__global__ void mask0_kernel(const float* __restrict__ x, float* __restrict__ m0, int N) {
  int p = blockIdx.x * blockDim.x + threadIdx.x;
  if (p >= N) return;
  int b = p / (H0 * W0);
  int r = p % (H0 * W0);
  bool any = false;
  #pragma unroll
  for (int c = 0; c < 6; ++c)
    any |= (x[(b * 6 + c) * (H0 * W0) + r] != 0.0f);
  m0[p] = any ? 1.0f : 0.0f;
}

// ------------------- WMMA implicit-GEMM 3x3 SAME conv ----------------------
// One wave computes a 16-pixel x 16-channel tile: D = A(16xK) * B(Kx16) + bias.
// A rows = im2col of 16 linearized pixels (row-major in K -> per-lane fragment
// is two contiguous 16B runs). B staged TRANSPOSED (N-major) so the per-lane
// fragment (same K mapping as A at fixed N) is two contiguous 16B runs.
// A frag (ISA 7.12.2): lane l, row M=l&15, elems 0..7 -> K=kbase+j,
//   elems 8..15 -> K=kbase+16+(j-8), kbase=(l<16?0:8).
// B frag: lane l, col N=l&15, same K mapping as A (kbase=(l<16?0:8)).
template <int CIN, int COUT, bool LRELU, bool NCHW_IN, bool NCHW_OUT>
__global__ void conv3x3_wmma_kernel(const float* __restrict__ in,
                                    const float* __restrict__ wgt,
                                    const float* __restrict__ bias,
                                    const float* __restrict__ mask,
                                    float* __restrict__ out,
                                    int B, int H, int W) {
  constexpr int KRAW   = 9 * CIN;
  constexpr int KPAD   = ((KRAW + 31) / 32) * 32;
  constexpr int KLDS   = KPAD + 8;            // +16B row pad (keeps 16B align)
  constexpr int NTILES = (COUT + 15) / 16;
  constexpr int NPAD   = NTILES * 16;
  constexpr int WAVES  = 2;

  __shared__ alignas(16) _Float16 As[WAVES][16][KLDS];
  __shared__ alignas(16) _Float16 Bs[WAVES][NPAD][KLDS];   // transposed: [N][K]

  const int wave = threadIdx.x >> 5;
  const int lane = threadIdx.x & 31;
  const int total = B * H * W;
  const int p0 = (blockIdx.x * WAVES + wave) * 16;

  // ---- stage A tile (im2col of 16 pixels), packed 2xf16 per store ----
  for (int idx = lane; idx < 16 * (KPAD / 2); idx += 32) {
    int m = idx / (KPAD / 2);
    int k0 = (idx % (KPAD / 2)) * 2;
    int p = p0 + m;
    float v0 = 0.0f, v1 = 0.0f;
    if (p < total) {
      int b = p / (H * W);
      int r = p % (H * W);
      int h = r / W, px = r % W;
      #pragma unroll
      for (int s = 0; s < 2; ++s) {
        int k = k0 + s;
        if (k < KRAW) {
          int tap = k / CIN, c = k % CIN;
          int hh = h + tap / 3 - 1, ww = px + tap % 3 - 1;
          if (hh >= 0 && hh < H && ww >= 0 && ww < W) {
            float v = NCHW_IN ? in[((b * CIN + c) * H + hh) * W + ww]
                              : in[((b * H + hh) * W + ww) * CIN + c];
            if (s == 0) v0 = v; else v1 = v;
          }
        }
      }
    }
    *reinterpret_cast<unsigned*>(&As[wave][m][k0]) = pack2h(v0, v1);
  }
  // ---- stage B tile transposed (weights HWIO: flat k*COUT + n) ----
  for (int idx = lane; idx < NPAD * (KPAD / 2); idx += 32) {
    int n = idx / (KPAD / 2);
    int k0 = (idx % (KPAD / 2)) * 2;
    float v0 = (k0 < KRAW && n < COUT) ? wgt[k0 * COUT + n] : 0.0f;
    float v1 = (k0 + 1 < KRAW && n < COUT) ? wgt[(k0 + 1) * COUT + n] : 0.0f;
    *reinterpret_cast<unsigned*>(&Bs[wave][n][k0]) = pack2h(v0, v1);
  }
  __syncthreads();

  const int kbaseA = (lane < 16) ? 0 : 8;
  const int kbaseB = (lane < 16) ? 0 : 8;
  const int mrow = lane & 15;

  for (int nt = 0; nt < NTILES; ++nt) {
    v8f acc = {};
    const int ncol = nt * 16 + mrow;
    #pragma unroll
    for (int kc = 0; kc < KPAD / 32; ++kc) {
      v16h a, bm;
      // A fragment: two contiguous 16B runs of row mrow
      v8h* ap = reinterpret_cast<v8h*>(&a);
      ap[0] = *reinterpret_cast<const v8h*>(&As[wave][mrow][kc * 32 + kbaseA]);
      ap[1] = *reinterpret_cast<const v8h*>(&As[wave][mrow][kc * 32 + kbaseA + 16]);
      // B fragment: two contiguous 16B runs of column ncol (stored N-major)
      v8h* bp = reinterpret_cast<v8h*>(&bm);
      bp[0] = *reinterpret_cast<const v8h*>(&Bs[wave][ncol][kc * 32 + kbaseB]);
      bp[1] = *reinterpret_cast<const v8h*>(&Bs[wave][ncol][kc * 32 + kbaseB + 16]);
      acc = __builtin_amdgcn_wmma_f32_16x16x32_f16(
          /*neg_a=*/false, a, /*neg_b=*/false, bm,
          /*c_mod=*/(short)0, acc, /*reuse_a=*/false, /*reuse_b=*/false);
    }
    // ---- epilogue: bias + leaky-relu + mask, store (C/D layout 7.12.2) ----
    #pragma unroll
    for (int r = 0; r < 8; ++r) {
      int M = (lane < 16) ? r : r + 8;
      int p = p0 + M;
      int n = ncol;
      if (p < total && n < COUT) {
        float v = acc[r] + bias[n];
        if (LRELU) v = (v > 0.0f) ? v : 0.01f * v;
        v = (mask[p] != 0.0f) ? v : 0.0f;
        if (NCHW_OUT) {
          int b = p / (H * W);
          int rr = p % (H * W);
          out[((b * COUT + n) * H + rr / W) * W + (rr % W)] = v;
        } else {
          out[p * COUT + n] = v;
        }
      }
    }
  }
}

// ------------------------------ max pools ----------------------------------
template <int KH, int KW>
__global__ void maskpool_kernel(const float* __restrict__ mi, float* __restrict__ mo,
                                int B, int Ho, int Wo) {
  int p = blockIdx.x * blockDim.x + threadIdx.x;
  int N = B * Ho * Wo;
  if (p >= N) return;
  int b = p / (Ho * Wo);
  int r = p % (Ho * Wo);
  int ho = r / Wo, wo = r % Wo;
  int Hin = Ho * KH, Win = Wo * KW;
  float v = 0.0f;
  for (int i = 0; i < KH; ++i)
    for (int j = 0; j < KW; ++j)
      v = fmaxf(v, mi[(b * Hin + ho * KH + i) * Win + wo * KW + j]);
  mo[p] = (v > 0.0f) ? 1.0f : 0.0f;
}

template <int KH, int KW, int C>
__global__ void pool_kernel(const float* __restrict__ in, const float* __restrict__ mi,
                            const float* __restrict__ mo, float* __restrict__ out,
                            int B, int Ho, int Wo) {
  int idx = blockIdx.x * blockDim.x + threadIdx.x;
  int N = B * Ho * Wo * C;
  if (idx >= N) return;
  int c = idx % C;
  int t = idx / C;
  int wo = t % Wo; t /= Wo;
  int ho = t % Ho;
  int b = t / Ho;
  int Hin = Ho * KH, Win = Wo * KW;
  float best = -1e30f;
  for (int i = 0; i < KH; ++i)
    for (int j = 0; j < KW; ++j) {
      int pi = (b * Hin + ho * KH + i) * Win + wo * KW + j;
      if (mi[pi] != 0.0f) best = fmaxf(best, in[pi * C + c]);
    }
  int pm = (b * Ho + ho) * Wo + wo;
  out[idx] = (mo[pm] != 0.0f) ? best : 0.0f;
}

// --------------------------- latent head -----------------------------------
__global__ void latent_kernel(const float* __restrict__ y,   // (B,4,16,32)
                              const float* __restrict__ wmu, const float* __restrict__ bmu,
                              const float* __restrict__ wlv, const float* __restrict__ blv,
                              const float* __restrict__ m3,  const float* __restrict__ eps,
                              float* __restrict__ mu_o, float* __restrict__ lv_o,
                              float* __restrict__ z0, float* __restrict__ stdv, int NE) {
  int e = blockIdx.x * blockDim.x + threadIdx.x;
  if (e >= NE) return;
  float msk = m3[e];
  #pragma unroll
  for (int i = 0; i < LATD; ++i) {
    float mu = bmu[i], lv = blv[i];
    for (int c = 0; c < 32; ++c) {
      float yv = y[e * 32 + c];
      mu += yv * wmu[i * 32 + c];
      lv += yv * wlv[i * 32 + c];
    }
    mu = (msk != 0.0f) ? mu : 0.0f;
    lv = (msk != 0.0f) ? lv : 0.0f;
    float st = __expf(0.5f * lv);
    mu_o[e * LATD + i] = mu;
    lv_o[e * LATD + i] = lv;
    stdv[e * LATD + i] = st;
    z0[e * LATD + i] = mu + eps[e * LATD + i] * st;
  }
}

// ------------------ rational-quadratic-spline (one dim) --------------------
__device__ __forceinline__ void rqs_one(float x, const float* uw, const float* uh,
                                        const float* ud, float& yout, float& ldout) {
  const float BD = 5.0f;
  float mw = fmaxf(fmaxf(uw[0], uw[1]), fmaxf(uw[2], uw[3]));
  float mh = fmaxf(fmaxf(uh[0], uh[1]), fmaxf(uh[2], uh[3]));
  float ew[4], eh[4], sw = 0.0f, sh = 0.0f;
  #pragma unroll
  for (int k = 0; k < 4; ++k) { ew[k] = __expf(uw[k] - mw); sw += ew[k];
                                eh[k] = __expf(uh[k] - mh); sh += eh[k]; }
  float xk[5], yk[5];
  xk[0] = -BD; yk[0] = -BD;
  #pragma unroll
  for (int k = 0; k < 4; ++k) {
    xk[k + 1] = xk[k] + ew[k] * (2.0f * BD) / sw;
    yk[k + 1] = yk[k] + eh[k] * (2.0f * BD) / sh;
  }
  float dk[5];
  dk[0] = 1.0f; dk[4] = 1.0f;
  #pragma unroll
  for (int k = 0; k < 3; ++k) {
    float v = ud[k];
    dk[k + 1] = (v > 20.0f) ? v : log1pf(__expf(v));   // softplus
  }
  bool inside = (x >= -BD) && (x <= BD);
  float xc = fminf(fmaxf(x, -BD), BD);
  int idx = 0;
  #pragma unroll
  for (int k = 1; k < 4; ++k) if (xc >= xk[k]) idx = k;
  float x0 = xk[idx], x1 = xk[idx + 1];
  float y0 = yk[idx], y1 = yk[idx + 1];
  float d0 = dk[idx], d1 = dk[idx + 1];
  float dx = x1 - x0, dy = y1 - y0;
  float s = dy / dx;
  float tt = (xc - x0) / dx, omt = 1.0f - tt;
  float den = s + (d1 + d0 - 2.0f * s) * tt * omt;
  float yv = y0 + dy * (s * tt * tt + d0 * tt * omt) / den;
  float ld = __logf(s * s * (d1 * tt * tt + 2.0f * s * tt * omt + d0 * omt * omt))
             - 2.0f * __logf(den);
  yout = inside ? yv : x;
  ldout = inside ? ld : 0.0f;
}

// --------------------- masked autoregressive NSF flow ----------------------
__global__ void flow_kernel(const float* __restrict__ z0buf, const float* __restrict__ stdbuf,
                            const float* __restrict__ eps,
                            const float* __restrict__ fW1, const float* __restrict__ fb1,
                            const float* __restrict__ fW2, const float* __restrict__ fb2,
                            const float* __restrict__ fW3, const float* __restrict__ fb3,
                            const float* __restrict__ m3,
                            float* __restrict__ zf_out, float* __restrict__ zf_ws,
                            float* __restrict__ kpc, int NE) {
  int e = blockIdx.x * blockDim.x + threadIdx.x;
  if (e >= NE) return;
  float z[LATD];
  #pragma unroll
  for (int i = 0; i < LATD; ++i) z[i] = z0buf[e * LATD + i];
  float ld_tot = 0.0f;

  for (int t = 0; t < NFLOW; ++t) {
    const float* W1 = fW1 + t * HIDD * LATD;
    const float* B1 = fb1 + t * HIDD;
    const float* W2 = fW2 + t * HIDD * HIDD;
    const float* B2 = fb2 + t * HIDD;
    const float* W3 = fW3 + t * LATD * NPP * HIDD;
    const float* B3 = fb3 + t * LATD * NPP;
    float h1[HIDD], h2[HIDD], pp[LATD * NPP];
    // layer 1: M1[i][j] = (i%3+1) >= (j+1)
    for (int i = 0; i < HIDD; ++i) {
      int hdi = i % 3 + 1;
      float a = B1[i];
      #pragma unroll
      for (int j = 0; j < LATD; ++j)
        if (hdi >= j + 1) a += W1[i * LATD + j] * z[j];
      h1[i] = fmaxf(a, 0.0f);
    }
    // layer 2: M2[i][j] = (i%3+1) >= (j%3+1)
    for (int i = 0; i < HIDD; ++i) {
      int hdi = i % 3 + 1;
      float a = B2[i];
      for (int j = 0; j < HIDD; ++j)
        if (hdi >= j % 3 + 1) a += W2[i * HIDD + j] * h1[j];
      h2[i] = fmaxf(a, 0.0f);
    }
    // layer 3: M3[r][j] = (r/11+1) > (j%3+1)
    for (int r = 0; r < LATD * NPP; ++r) {
      int rd = r / NPP + 1;
      float a = B3[r];
      for (int j = 0; j < HIDD; ++j)
        if (rd > j % 3 + 1) a += W3[r * HIDD + j] * h2[j];
      pp[r] = a;
    }
    float zn[LATD];
    #pragma unroll
    for (int l = 0; l < LATD; ++l) {
      float yv, ld;
      rqs_one(z[l], pp + l * NPP, pp + l * NPP + 4, pp + l * NPP + 8, yv, ld);
      zn[l] = yv;
      ld_tot += ld;
    }
    if (t < NFLOW - 1) {
      #pragma unroll
      for (int l = 0; l < LATD; ++l) z[l] = zn[LATD - 1 - l];  // reverse
    } else {
      #pragma unroll
      for (int l = 0; l < LATD; ++l) z[l] = zn[l];
    }
  }
  // KLD per cell: log_q0 - log_det - log_p ; (z0-mu)/std == eps exactly
  const float HALF_LOG2PI = 0.91893853320467267f;
  float lq0 = 0.0f, lp = 0.0f;
  #pragma unroll
  for (int i = 0; i < LATD; ++i) {
    float ei = eps[e * LATD + i];
    float st = stdbuf[e * LATD + i];
    lq0 += -0.5f * ei * ei - __logf(st) - HALF_LOG2PI;
    lp  += -0.5f * z[i] * z[i] - HALF_LOG2PI;
  }
  float msk = m3[e];
  kpc[e] = (msk != 0.0f) ? (lq0 - ld_tot - lp) : 0.0f;
  #pragma unroll
  for (int i = 0; i < LATD; ++i) {
    float v = (msk != 0.0f) ? z[i] : 0.0f;
    zf_out[e * LATD + i] = v;
    zf_ws[e * LATD + i] = v;
  }
}

// ------------------------- KLD per-batch reduce ----------------------------
__global__ void kld_reduce_kernel(const float* __restrict__ kpc, float* __restrict__ kld) {
  __shared__ float s[64];
  int b = blockIdx.x, t = threadIdx.x;
  s[t] = kpc[b * 64 + t];
  __syncthreads();
  for (int off = 32; off > 0; off >>= 1) {
    if (t < off) s[t] += s[t + off];
    __syncthreads();
  }
  if (t == 0) kld[b] = s[0];
}

// ------------------------ decoder linear (4 -> 32) -------------------------
__global__ void dec_kernel(const float* __restrict__ zf, const float* __restrict__ wlin,
                           const float* __restrict__ blin, const float* __restrict__ m3,
                           float* __restrict__ out, int NE) {
  int idx = blockIdx.x * blockDim.x + threadIdx.x;
  if (idx >= NE * 32) return;
  int c = idx % 32, e = idx / 32;
  float a = blin[c];
  #pragma unroll
  for (int i = 0; i < LATD; ++i) a += zf[e * LATD + i] * wlin[c * LATD + i];
  out[idx] = (m3[e] != 0.0f) ? a : 0.0f;
}

// ---------------------- block-upsampling "invconv" -------------------------
// out[b, h*KH+i, w*KW+j, d] = sum_c in[b,h,w,c] * u[i,j,c,d] + bu[d] (masked mo)
template <int KH, int KW, int C>
__global__ void invconv_kernel(const float* __restrict__ in, const float* __restrict__ u,
                               const float* __restrict__ bu, const float* __restrict__ mo,
                               float* __restrict__ out, int B, int Hin, int Win) {
  int idx = blockIdx.x * blockDim.x + threadIdx.x;
  int Ho = Hin * KH, Wo = Win * KW;
  int N = B * Ho * Wo * C;
  if (idx >= N) return;
  int d = idx % C;
  int t = idx / C;
  int wo = t % Wo; t /= Wo;
  int ho = t % Ho;
  int b = t / Ho;
  int h = ho / KH, i = ho % KH, w = wo / KW, j = wo % KW;
  float a = bu[d];
  for (int c = 0; c < C; ++c)
    a += in[((b * Hin + h) * Win + w) * C + c] * u[((i * KW + j) * C + c) * C + d];
  int pm = (b * Ho + ho) * Wo + wo;
  out[idx] = (mo[pm] != 0.0f) ? a : 0.0f;
}

// ---------------------------------------------------------------------------
extern "C" void kernel_launch(void* const* d_in, const int* in_sizes, int n_in,
                              void* d_out, int out_size, void* d_ws, size_t ws_size,
                              hipStream_t stream) {
  const float* x    = (const float*)d_in[0];
  const float* eps  = (const float*)d_in[1];
  const float* w1   = (const float*)d_in[2];
  const float* b1   = (const float*)d_in[3];
  const float* w2   = (const float*)d_in[4];
  const float* b2   = (const float*)d_in[5];
  const float* w3   = (const float*)d_in[6];
  const float* b3   = (const float*)d_in[7];
  const float* wmu  = (const float*)d_in[8];
  const float* bmu  = (const float*)d_in[9];
  const float* wlv  = (const float*)d_in[10];
  const float* blv  = (const float*)d_in[11];
  const float* wlin = (const float*)d_in[12];
  const float* blin = (const float*)d_in[13];
  const float* u1   = (const float*)d_in[14];
  const float* bu1  = (const float*)d_in[15];
  const float* w4   = (const float*)d_in[16];
  const float* b4   = (const float*)d_in[17];
  const float* u2   = (const float*)d_in[18];
  const float* bu2  = (const float*)d_in[19];
  const float* w5   = (const float*)d_in[20];
  const float* b5   = (const float*)d_in[21];
  const float* u3   = (const float*)d_in[22];
  const float* bu3  = (const float*)d_in[23];
  const float* w6   = (const float*)d_in[24];
  const float* b6   = (const float*)d_in[25];
  const float* fW1  = (const float*)d_in[26];
  const float* fb1  = (const float*)d_in[27];
  const float* fW2  = (const float*)d_in[28];
  const float* fb2  = (const float*)d_in[29];
  const float* fW3  = (const float*)d_in[30];
  const float* fb3  = (const float*)d_in[31];

  // ---- workspace layout (floats) ----
  float* ws = (float*)d_ws;
  float* m0 = ws;                        // 128*200*128
  float* m1 = m0 + BN * H0 * W0;         // 128*100*64
  float* m2 = m1 + BN * H1 * W1;         // 128*20*32
  float* m3 = m2 + BN * H2 * W2;         // 128*4*16
  float* bufA = m3 + BN * H3 * W3;       // (B,200,128,8)  y1 / up3
  float* bufB = bufA + BN * H0 * W0 * 8; // (B,100,64,8)   y1p / c5
  float* bufC = bufB + BN * H1 * W1 * 8; // (B,100,64,16)  y2 / up2
  float* bufD = bufC + BN * H1 * W1 * 16;// (B,20,32,16)   y2p / c4
  float* bufE = bufD + BN * H2 * W2 * 16;// (B,20,32,32)   y3 / up1
  float* bufF = bufE + BN * H2 * W2 * 32;// (B,4,16,32)    y3p / dec
  float* z0   = bufF + BN * H3 * W3 * 32;// 8192*4
  float* stdv = z0 + BN * H3 * W3 * LATD;
  float* zfws = stdv + BN * H3 * W3 * LATD;
  float* kpc  = zfws + BN * H3 * W3 * LATD; // 8192

  // ---- output layout ----
  float* out   = (float*)d_out;                 // (B,6,200,128)
  float* mu_o  = out + BN * 6 * H0 * W0;        // 32768
  float* lv_o  = mu_o + BN * H3 * W3 * LATD;    // 32768
  float* kld_o = lv_o + BN * H3 * W3 * LATD;    // 128
  float* zf_o  = kld_o + BN;                    // 32768

  const int NE = BN * H3 * W3;   // 8192 latent cells
  const int T256 = 256;

  // ========================= encoder =========================
  {
    int N = BN * H0 * W0;
    mask0_kernel<<<CDIV(N, T256), T256, 0, stream>>>(x, m0, N);
    conv3x3_wmma_kernel<6, 8, true, true, false>
        <<<CDIV(N, 32), 64, 0, stream>>>(x, w1, b1, m0, bufA, BN, H0, W0);
  }
  {
    int Nm = BN * H1 * W1;
    maskpool_kernel<2, 2><<<CDIV(Nm, T256), T256, 0, stream>>>(m0, m1, BN, H1, W1);
    pool_kernel<2, 2, 8><<<CDIV(Nm * 8, T256), T256, 0, stream>>>(bufA, m0, m1, bufB, BN, H1, W1);
    conv3x3_wmma_kernel<8, 16, true, false, false>
        <<<CDIV(Nm, 32), 64, 0, stream>>>(bufB, w2, b2, m1, bufC, BN, H1, W1);
  }
  {
    int Nm = BN * H2 * W2;
    maskpool_kernel<5, 2><<<CDIV(Nm, T256), T256, 0, stream>>>(m1, m2, BN, H2, W2);
    pool_kernel<5, 2, 16><<<CDIV(Nm * 16, T256), T256, 0, stream>>>(bufC, m1, m2, bufD, BN, H2, W2);
    conv3x3_wmma_kernel<16, 32, true, false, false>
        <<<CDIV(Nm, 32), 64, 0, stream>>>(bufD, w3, b3, m2, bufE, BN, H2, W2);
  }
  {
    int Nm = BN * H3 * W3;
    maskpool_kernel<5, 2><<<CDIV(Nm, T256), T256, 0, stream>>>(m2, m3, BN, H3, W3);
    pool_kernel<5, 2, 32><<<CDIV(Nm * 32, T256), T256, 0, stream>>>(bufE, m2, m3, bufF, BN, H3, W3);
  }

  // ======================= latent + flow =====================
  latent_kernel<<<CDIV(NE, T256), T256, 0, stream>>>(bufF, wmu, bmu, wlv, blv, m3, eps,
                                                     mu_o, lv_o, z0, stdv, NE);
  flow_kernel<<<CDIV(NE, 128), 128, 0, stream>>>(z0, stdv, eps, fW1, fb1, fW2, fb2,
                                                 fW3, fb3, m3, zf_o, zfws, kpc, NE);
  kld_reduce_kernel<<<BN, 64, 0, stream>>>(kpc, kld_o);

  // ========================= decoder =========================
  dec_kernel<<<CDIV(NE * 32, T256), T256, 0, stream>>>(zfws, wlin, blin, m3, bufF, NE);
  {
    int N = BN * H2 * W2 * 32;
    invconv_kernel<5, 2, 32><<<CDIV(N, T256), T256, 0, stream>>>(bufF, u1, bu1, m2, bufE, BN, H3, W3);
    conv3x3_wmma_kernel<32, 16, false, false, false>
        <<<CDIV(BN * H2 * W2, 32), 64, 0, stream>>>(bufE, w4, b4, m2, bufD, BN, H2, W2);
  }
  {
    int N = BN * H1 * W1 * 16;
    invconv_kernel<5, 2, 16><<<CDIV(N, T256), T256, 0, stream>>>(bufD, u2, bu2, m1, bufC, BN, H2, W2);
    conv3x3_wmma_kernel<16, 8, false, false, false>
        <<<CDIV(BN * H1 * W1, 32), 64, 0, stream>>>(bufC, w5, b5, m1, bufB, BN, H1, W1);
  }
  {
    int N = BN * H0 * W0 * 8;
    invconv_kernel<2, 2, 8><<<CDIV(N, T256), T256, 0, stream>>>(bufB, u3, bu3, m0, bufA, BN, H1, W1);
    conv3x3_wmma_kernel<8, 6, false, false, true>
        <<<CDIV(BN * H0 * W0, 32), 64, 0, stream>>>(bufA, w6, b6, m0, out, BN, H0, W0);
  }
}
